// CrossExchangingLayer_1589137899525
// MI455X (gfx1250) — matrix operand order repaired and checked by reading, not compiled, not run on hardware
//
#include <hip/hip_runtime.h>
#include <hip/hip_bf16.h>

#define B_   32
#define L_   512
#define D_   256
#define H_   8
#define HD_  32
#define DFF_ 2048
#define BL_  (B_ * L_)

typedef __attribute__((ext_vector_type(16))) __bf16 v16bf;
typedef __attribute__((ext_vector_type(8)))  float  v8f;

__device__ __forceinline__ v8f vzero8() {
  v8f z;
#pragma unroll
  for (int i = 0; i < 8; ++i) z[i] = 0.f;
  return z;
}

__device__ __forceinline__ v8f wmma_bf16(v16bf a, v16bf b, v8f c) {
  // D = A(16x32 bf16) * B(32x16 bf16) + C(16x16 f32)
  return __builtin_amdgcn_wmma_f32_16x16x32_bf16(false, a, false, b, (short)0, c,
                                                 false, false);
}

// Raw LDS byte offset of a __shared__ object (addrspace(3) pointers are the
// wave-relative LDS byte address on amdgcn).
__device__ __forceinline__ unsigned lds_addr_of(const void* p) {
  return (unsigned)(unsigned long long)(__attribute__((address_space(3))) const char*)p;
}

// Async global->LDS 16B copy (GLOBAL_LOAD_ASYNC_TO_LDS_B128, ASYNCcnt-tracked).
__device__ __forceinline__ void async_copy16(unsigned ldsoff, unsigned long long gaddr) {
  asm volatile("global_load_async_to_lds_b128 %0, %1, off"
               :
               : "v"(ldsoff), "v"(gaddr)
               : "memory");
}
__device__ __forceinline__ void wait_async_le2() {
  asm volatile("s_wait_asynccnt 0x2" ::: "memory");
}
__device__ __forceinline__ void wait_async_le0() {
  asm volatile("s_wait_asynccnt 0x0" ::: "memory");
}

// A-matrix fragment (16x32, bf16): lanes 0-15 = rows, K pairs per VGPR.
// lane<16 holds K {0..15}; lane>=16 holds K {16..31}.
__device__ __forceinline__ v16bf frag_a(const __bf16* __restrict__ base, int stride,
                                        int lane) {
  const int r  = lane & 15;
  const int kb = (lane >> 4) << 4;
  const __bf16* p = base + (size_t)r * stride;
  v16bf a;
#pragma unroll
  for (int i = 0; i < 16; ++i) a[i] = p[kb + i];
  return a;
}

// B-matrix fragment from row-major [N][K] storage (weightsᵀ / K-matrix):
// lane n<16: col N=n, K=0..15 ; lane>=16: col N=n-16, K=16..31.
__device__ __forceinline__ v16bf frag_b_nt(const __bf16* __restrict__ base, int stride,
                                           int lane) {
  const int n  = lane & 15;
  const int kb = (lane >> 4) << 4;
  const __bf16* p = base + (size_t)n * stride + kb;
  v16bf b;
#pragma unroll
  for (int i = 0; i < 16; ++i) b[i] = p[i];
  return b;
}

// B-matrix fragment from row-major [K][N] storage (V-matrix, NN form).
__device__ __forceinline__ v16bf frag_b_nn(const __bf16* __restrict__ base, int stride,
                                           int lane) {
  const int n  = lane & 15;
  const int kb = (lane >> 4) << 4;
  const __bf16* p = base + (size_t)kb * stride + n;
  v16bf b;
#pragma unroll
  for (int i = 0; i < 16; ++i) b[i] = p[(size_t)i * stride];
  return b;
}

// ---------------------------------------------------------------------------
// Tiled WMMA GEMM:  C[M,N] = act(A[M,K] * Bw[N,K]^T + bias[N])
// Block 128x128, 256 threads = 8 waves (2 M x 4 N), wave tile 64x32 (4x2 WMMA).
// Tiles staged with GLOBAL_LOAD_ASYNC_TO_LDS_B128 into double-buffered LDS:
// tile i+1 is issued before computing tile i; per-wave s_wait_asynccnt 2
// (async loads complete in order) ensures tile i landed while i+1 stays in
// flight. A, Bw bf16; optional fp32 (Cf) and/or bf16 (Cb) outputs.
// M%128==N%128==K%32==0 for all call sites.
// ---------------------------------------------------------------------------
__global__ __launch_bounds__(256) void gemm_bf16(
    const __bf16* __restrict__ A, const __bf16* __restrict__ Bw,
    const float* __restrict__ bias, float* __restrict__ Cf,
    __bf16* __restrict__ Cb, int M, int N, int K, int relu) {
  __shared__ __bf16 Alds[2][128 * 40];  // +8 pad: rows stay 16B aligned, no conflicts
  __shared__ __bf16 Blds[2][128 * 40];
  const int tid  = threadIdx.x;
  const int lane = tid & 31;
  const int wid  = tid >> 5;
  const int wm   = wid >> 2;  // 0..1
  const int wn   = wid & 3;   // 0..3
  const int mBase = blockIdx.y * 128;
  const int nBase = blockIdx.x * 128;
  const int lrow  = tid >> 1;  // 0..127
  const int lhalf = tid & 1;   // which 16-bf16 half-row

  const unsigned aoff[2] = {lds_addr_of(&Alds[0][lrow * 40 + lhalf * 16]),
                            lds_addr_of(&Alds[1][lrow * 40 + lhalf * 16])};
  const unsigned boff[2] = {lds_addr_of(&Blds[0][lrow * 40 + lhalf * 16]),
                            lds_addr_of(&Blds[1][lrow * 40 + lhalf * 16])};
  const __bf16* gA = A + (size_t)(mBase + lrow) * K + lhalf * 16;
  const __bf16* gB = Bw + (size_t)(nBase + lrow) * K + lhalf * 16;

  v8f acc[4][2];
#pragma unroll
  for (int mi = 0; mi < 4; ++mi)
#pragma unroll
    for (int ni = 0; ni < 2; ++ni) acc[mi][ni] = vzero8();

  const int T = K >> 5;  // K/32 tiles
  // Prologue: tile 0 in flight.
  async_copy16(aoff[0], (unsigned long long)gA);
  async_copy16(boff[0], (unsigned long long)gB);

  for (int i = 0; i < T; ++i) {
    if (i + 1 < T) {  // issue next tile into the other buffer, keep it in flight
      const int nb = (i + 1) & 1;
      async_copy16(aoff[nb], (unsigned long long)(gA + (size_t)(i + 1) * 32));
      async_copy16(boff[nb], (unsigned long long)(gB + (size_t)(i + 1) * 32));
      wait_async_le2();  // tile i's 2 loads done (in-order completion)
    } else {
      wait_async_le0();
    }
    __syncthreads();  // tile i visible to all waves

    const __bf16* Ab = &Alds[i & 1][0];
    const __bf16* Bb = &Blds[i & 1][0];
    v16bf af[4], bf[2];
#pragma unroll
    for (int mi = 0; mi < 4; ++mi)
      af[mi] = frag_a(Ab + (wm * 64 + mi * 16) * 40, 40, lane);
#pragma unroll
    for (int ni = 0; ni < 2; ++ni)
      bf[ni] = frag_b_nt(Bb + (wn * 32 + ni * 16) * 40, 40, lane);
#pragma unroll
    for (int mi = 0; mi < 4; ++mi)
#pragma unroll
      for (int ni = 0; ni < 2; ++ni)
        acc[mi][ni] = wmma_bf16(af[mi], bf[ni], acc[mi][ni]);

    __syncthreads();  // all reads of buf[i&1] done before it is overwritten
  }

  const int nl   = lane & 15;
  const int madd = (lane >> 4) << 3;
#pragma unroll
  for (int mi = 0; mi < 4; ++mi)
#pragma unroll
    for (int ni = 0; ni < 2; ++ni)
#pragma unroll
      for (int v = 0; v < 8; ++v) {
        const int row = mBase + wm * 64 + mi * 16 + madd + v;
        const int col = nBase + wn * 32 + ni * 16 + nl;
        float val = acc[mi][ni][v] + bias[col];
        if (relu) val = fmaxf(val, 0.f);
        if (Cf) Cf[(size_t)row * N + col] = val;
        if (Cb) Cb[(size_t)row * N + col] = (__bf16)val;
      }
}

// ---------------------------------------------------------------------------
// Fused flash attention per (b,h): grid (B*H, L/64), 128 threads = 4 waves,
// each wave owns 16 query rows. HD=32 == WMMA K, so one WMMA per 16x16 score
// tile. Online softmax in fp32 with 16-lane shuffles; P transposed to A-layout
// through a per-wave LDS scratch; K/V streamed straight from L2 (qkv ~25MB).
// ---------------------------------------------------------------------------
__global__ __launch_bounds__(128) void attn_fused(const __bf16* __restrict__ qkv,
                                                  __bf16* __restrict__ outp) {
  __shared__ __bf16 Pl[4][16][32];
  const int lane = threadIdx.x & 31;
  const int w    = threadIdx.x >> 5;
  const int b    = blockIdx.x >> 3;
  const int h    = blockIdx.x & 7;
  const int q0   = blockIdx.y * 64 + w * 16;
  const float scale = 0.17677669529663687f;  // 1/sqrt(HD)

  const __bf16* qb = qkv + ((size_t)(b * L_) + q0) * (3 * D_) + h * HD_;
  const __bf16* kb = qkv + (size_t)(b * L_) * (3 * D_) + D_ + h * HD_;
  const __bf16* vb = qkv + (size_t)(b * L_) * (3 * D_) + 2 * D_ + h * HD_;

  const v16bf qf = frag_a(qb, 3 * D_, lane);

  v8f m, l, O0 = vzero8(), O1 = vzero8();
#pragma unroll
  for (int v = 0; v < 8; ++v) {
    m[v] = -3.0e38f;
    l[v] = 0.f;
  }
  const int nl   = lane & 15;
  const int madd = (lane >> 4) << 3;

  for (int j = 0; j < L_; j += 32) {
    v16bf kf0 = frag_b_nt(kb + (size_t)j * (3 * D_), 3 * D_, lane);
    v16bf kf1 = frag_b_nt(kb + (size_t)(j + 16) * (3 * D_), 3 * D_, lane);
    v8f s0 = wmma_bf16(qf, kf0, vzero8());
    v8f s1 = wmma_bf16(qf, kf1, vzero8());
#pragma unroll
    for (int v = 0; v < 8; ++v) {
      const float a0 = s0[v] * scale;
      const float a1 = s1[v] * scale;
      float t = fmaxf(a0, a1);
#pragma unroll
      for (int off = 1; off < 16; off <<= 1) t = fmaxf(t, __shfl_xor(t, off, 32));
      const float mn = fmaxf(m[v], t);
      const float al = __expf(m[v] - mn);
      const float p0 = __expf(a0 - mn);
      const float p1 = __expf(a1 - mn);
      float ps = p0 + p1;
#pragma unroll
      for (int off = 1; off < 16; off <<= 1) ps += __shfl_xor(ps, off, 32);
      l[v] = l[v] * al + ps;
      O0[v] *= al;
      O1[v] *= al;
      m[v] = mn;
      Pl[w][madd + v][nl]      = (__bf16)p0;
      Pl[w][madd + v][16 + nl] = (__bf16)p1;
    }
    asm volatile("s_wait_dscnt 0x0" ::: "memory");  // wave-local LDS RAW
    v16bf pf  = frag_a(&Pl[w][0][0], 32, lane);
    v16bf vf0 = frag_b_nn(vb + (size_t)j * (3 * D_), 3 * D_, lane);
    v16bf vf1 = frag_b_nn(vb + (size_t)j * (3 * D_) + 16, 3 * D_, lane);
    O0 = wmma_bf16(pf, vf0, O0);
    O1 = wmma_bf16(pf, vf1, O1);
  }
#pragma unroll
  for (int v = 0; v < 8; ++v) {
    const float inv = 1.f / l[v];
    const size_t row = (size_t)(b * L_) + q0 + madd + v;
    outp[row * D_ + h * HD_ + nl]      = (__bf16)(O0[v] * inv);
    outp[row * D_ + h * HD_ + 16 + nl] = (__bf16)(O1[v] * inv);
  }
}

// Head-mean softmax row for q=0 (feeds the cross-exchange mask).
__global__ __launch_bounds__(512) void aw0_kernel(const __bf16* __restrict__ qkv,
                                                  float* __restrict__ aw0) {
  __shared__ float red[512];
  const int b = blockIdx.x >> 3;
  const int h = blockIdx.x & 7;
  const int j = threadIdx.x;
  const float scale = 0.17677669529663687f;
  const __bf16* q = qkv + (size_t)(b * L_) * (3 * D_) + h * HD_;
  const __bf16* k = qkv + ((size_t)(b * L_) + j) * (3 * D_) + D_ + h * HD_;
  float s = 0.f;
#pragma unroll
  for (int d = 0; d < HD_; ++d) s += (float)q[d] * (float)k[d];
  s *= scale;
  red[j] = s;
  __syncthreads();
  for (int st = 256; st > 0; st >>= 1) {
    if (j < st) red[j] = fmaxf(red[j], red[j + st]);
    __syncthreads();
  }
  const float mx = red[0];
  __syncthreads();
  const float e = __expf(s - mx);
  red[j] = e;
  __syncthreads();
  for (int st = 256; st > 0; st >>= 1) {
    if (j < st) red[j] += red[j + st];
    __syncthreads();
  }
  atomicAdd(&aw0[(size_t)b * L_ + j], e / (red[0] * (float)H_));
}

__global__ void cvt_bf16(const float* __restrict__ x, __bf16* __restrict__ y, int n) {
  const int i = blockIdx.x * blockDim.x + threadIdx.x;
  if (i < n) y[i] = (__bf16)x[i];
}

__global__ void zero_f32(float* __restrict__ x, int n) {
  const int i = blockIdx.x * blockDim.x + threadIdx.x;
  if (i < n) x[i] = 0.f;
}

// y = LN(x + r)*g + be ; optional fp32 and bf16 outputs. One row per block.
__global__ __launch_bounds__(256) void add_ln(const float* __restrict__ x,
                                              const float* __restrict__ r,
                                              const float* __restrict__ g,
                                              const float* __restrict__ be,
                                              float* __restrict__ yf,
                                              __bf16* __restrict__ yb) {
  __shared__ float red[256];
  const int row = blockIdx.x;
  const int t   = threadIdx.x;
  const size_t idx = (size_t)row * D_ + t;
  const float v = x[idx] + r[idx];
  red[t] = v;
  __syncthreads();
  for (int st = 128; st > 0; st >>= 1) {
    if (t < st) red[t] += red[t + st];
    __syncthreads();
  }
  const float mean = red[0] * (1.f / D_);
  __syncthreads();
  const float d = v - mean;
  red[t] = d * d;
  __syncthreads();
  for (int st = 128; st > 0; st >>= 1) {
    if (t < st) red[t] += red[t + st];
    __syncthreads();
  }
  const float y = d * rsqrtf(red[0] * (1.f / D_) + 1e-5f) * g[t] + be[t];
  if (yf) yf[idx] = y;
  if (yb) yb[idx] = (__bf16)y;
}

__global__ __launch_bounds__(256) void colmean(const float* __restrict__ x,
                                               float* __restrict__ mean) {
  const int b = blockIdx.x;
  const int d = threadIdx.x;
  float s = 0.f;
  for (int l = 0; l < L_; ++l) s += x[((size_t)b * L_ + l) * D_ + d];
  mean[(size_t)b * D_ + d] = s * (1.f / L_);
}

// Exact 0.5-quantile of 511 points = order statistic 255 (jnp.quantile hits it
// exactly since 0.5*510 is integral). Count-based selection, ties handled.
__global__ __launch_bounds__(256) void median_k(const float* __restrict__ aw0,
                                                float* __restrict__ th) {
  __shared__ float c[L_ - 1];
  const int b = blockIdx.x;
  for (int i = threadIdx.x; i < L_ - 1; i += blockDim.x)
    c[i] = aw0[(size_t)b * L_ + 1 + i];
  __syncthreads();
  const int kth = (L_ - 2) / 2;  // 255
  for (int i = threadIdx.x; i < L_ - 1; i += blockDim.x) {
    const float v = c[i];
    int lt = 0, eq = 0;
    for (int t = 0; t < L_ - 1; ++t) {
      const float u = c[t];
      lt += (u < v);
      eq += (u == v);
    }
    if (lt <= kth && kth < lt + eq) th[b] = v;
  }
}

__global__ __launch_bounds__(256) void exchange_k(float* __restrict__ x,
                                                  __bf16* __restrict__ xbv,
                                                  const float* __restrict__ aw0,
                                                  const float* __restrict__ th,
                                                  const float* __restrict__ om) {
  const int b = blockIdx.x / (L_ - 1);
  const int j = blockIdx.x % (L_ - 1);
  const int d = threadIdx.x;
  if (aw0[(size_t)b * L_ + 1 + j] < th[b]) {
    const float v = om[(size_t)b * D_ + d];
    const size_t idx = ((size_t)b * L_ + 1 + j) * D_ + d;
    x[idx]   = v;
    xbv[idx] = (__bf16)v;
  }
}

// ---------------------------------------------------------------------------
extern "C" void kernel_launch(void* const* d_in, const int* in_sizes, int n_in,
                              void* d_out, int out_size, void* d_ws, size_t ws_size,
                              hipStream_t stream) {
  const float* src[2] = {(const float*)d_in[0], (const float*)d_in[1]};
  const float* W_in  = (const float*)d_in[4];
  const float* b_in  = (const float*)d_in[5];
  const float* W_out = (const float*)d_in[6];
  const float* b_out = (const float*)d_in[7];
  const float* W1    = (const float*)d_in[8];
  const float* b1    = (const float*)d_in[9];
  const float* W2    = (const float*)d_in[10];
  const float* b2    = (const float*)d_in[11];
  const float* g1    = (const float*)d_in[12];
  const float* be1   = (const float*)d_in[13];
  const float* g2    = (const float*)d_in[14];
  const float* be2   = (const float*)d_in[15];
  float* out = (float*)d_out;

  char* p = (char*)d_ws;
  auto carve = [&](size_t bytes) -> char* {
    char* q = p;
    p += (bytes + 255) & ~(size_t)255;
    return q;
  };
  __bf16* wbf_in  = (__bf16*)carve((size_t)3 * D_ * D_ * 2);
  __bf16* wbf_out = (__bf16*)carve((size_t)D_ * D_ * 2);
  __bf16* wbf_1   = (__bf16*)carve((size_t)DFF_ * D_ * 2);
  __bf16* wbf_2   = (__bf16*)carve((size_t)D_ * DFF_ * 2);
  __bf16* xbf     = (__bf16*)carve((size_t)BL_ * D_ * 2);
  __bf16* qkv_bf  = (__bf16*)carve((size_t)BL_ * 3 * D_ * 2);
  __bf16* ao_bf   = (__bf16*)carve((size_t)BL_ * D_ * 2);
  float*  proj_f  = (float*)carve((size_t)BL_ * D_ * 4);
  float*  xf[2];
  __bf16* xb[2];
  float*  aw0[2];
  float*  xmean[2];
  float*  th[2];
  for (int s = 0; s < 2; ++s) {
    xf[s]    = (float*)carve((size_t)BL_ * D_ * 4);
    xb[s]    = (__bf16*)carve((size_t)BL_ * D_ * 2);
    aw0[s]   = (float*)carve((size_t)B_ * L_ * 4);
    xmean[s] = (float*)carve((size_t)B_ * D_ * 4);
    th[s]    = (float*)carve(256);
  }
  __bf16* hid_bf = (__bf16*)carve((size_t)BL_ * DFF_ * 2);
  float*  ff_f   = (float*)carve((size_t)BL_ * D_ * 4);

  auto cvt = [&](const float* a, __bf16* o, int n) {
    cvt_bf16<<<(n + 255) / 256, 256, 0, stream>>>(a, o, n);
  };
  cvt(W_in, wbf_in, 3 * D_ * D_);
  cvt(W_out, wbf_out, D_ * D_);
  cvt(W1, wbf_1, DFF_ * D_);
  cvt(W2, wbf_2, D_ * DFF_);

  // ---- MHA + first residual/LN, per stream -------------------------------
  for (int s = 0; s < 2; ++s) {
    cvt(src[s], xbf, BL_ * D_);
    gemm_bf16<<<dim3(3 * D_ / 128, BL_ / 128), 256, 0, stream>>>(
        xbf, wbf_in, b_in, nullptr, qkv_bf, BL_, 3 * D_, D_, 0);
    attn_fused<<<dim3(B_ * H_, L_ / 64), 128, 0, stream>>>(qkv_bf, ao_bf);
    gemm_bf16<<<dim3(D_ / 128, BL_ / 128), 256, 0, stream>>>(
        ao_bf, wbf_out, b_out, proj_f, nullptr, BL_, D_, D_, 0);
    zero_f32<<<(B_ * L_ + 255) / 256, 256, 0, stream>>>(aw0[s], B_ * L_);
    aw0_kernel<<<B_ * H_, 512, 0, stream>>>(qkv_bf, aw0[s]);
    add_ln<<<BL_, 256, 0, stream>>>(src[s], proj_f, g1, be1, xf[s], xb[s]);
  }

  // ---- Cross-exchange -----------------------------------------------------
  for (int s = 0; s < 2; ++s) {
    colmean<<<B_, 256, 0, stream>>>(xf[s], xmean[s]);
    median_k<<<B_, 256, 0, stream>>>(aw0[s], th[s]);
  }
  for (int s = 0; s < 2; ++s)
    exchange_k<<<B_ * (L_ - 1), 256, 0, stream>>>(xf[s], xb[s], aw0[s], th[s],
                                                  xmean[1 - s]);

  // ---- FFN + second residual/LN, per stream ------------------------------
  for (int s = 0; s < 2; ++s) {
    gemm_bf16<<<dim3(DFF_ / 128, BL_ / 128), 256, 0, stream>>>(
        xb[s], wbf_1, b1, nullptr, hid_bf, BL_, DFF_, D_, 1);
    gemm_bf16<<<dim3(D_ / 128, BL_ / 128), 256, 0, stream>>>(
        hid_bf, wbf_2, b2, ff_f, nullptr, BL_, D_, DFF_, 0);
    add_ln<<<BL_, 256, 0, stream>>>(xf[s], ff_f, g2, be2,
                                    out + (size_t)s * BL_ * D_, nullptr);
  }
  (void)in_sizes;
  (void)n_in;
  (void)out_size;
  (void)ws_size;
}
